// LearnableGlobalLocalMultiheadAttention_24816321036389
// MI455X (gfx1250) — matrix operand order repaired and checked by reading, not compiled, not run on hardware
//
#include <hip/hip_runtime.h>
#include <hip/hip_bf16.h>

// ---------------- problem constants ----------------
#define E_     1024
#define T_     1024
#define B_     4
#define H_     16
#define HD_    64
#define BH_    64       // B_*H_
#define NW_    9
#define MROWS  4096     // T_*B_
#define NPROJ  9216     // NW_*E_

typedef __attribute__((ext_vector_type(16))) __bf16 bf16x16;
typedef __attribute__((ext_vector_type(8)))  __bf16 bf16x8;
typedef __attribute__((ext_vector_type(8)))  float  f32x8;

// ---------------- helpers ----------------
static __device__ inline __bf16 f2bf(float f) {
    union { float f; unsigned u; } x; x.f = f;
    unsigned r = x.u + 0x7FFFu + ((x.u >> 16) & 1u);   // round-to-nearest-even
    unsigned short h = (unsigned short)(r >> 16);
    return __builtin_bit_cast(__bf16, h);
}

// CDNA5 async global->LDS copy (ASYNCcnt-tracked, no VGPR round trip).
// VDST = per-lane LDS byte address, VADDR = per-lane 64-bit global address.
static __device__ inline void async_copy_b128(const __bf16* g, __bf16* l) {
    unsigned lds_a = (unsigned)(reinterpret_cast<uintptr_t>(l));       // low 32 = LDS byte addr
    unsigned long long ga = (unsigned long long)(reinterpret_cast<uintptr_t>(g));
    asm volatile("global_load_async_to_lds_b128 %0, %1, off"
                 :: "v"(lds_a), "v"(ga) : "memory");
}
static __device__ inline void wait_async0() {
    asm volatile("s_wait_asynccnt 0" ::: "memory");
}

// A fragment (16x32 bf16, M x K): lanes 0-15 hold K=[0..7]+[16..23],
// lanes 16-31 hold K=[8..15]+[24..31] of rows M=lane%16.
static __device__ inline bf16x16 load_fragA(const __bf16* mat, long ld,
                                            long row0, long k0, int lane) {
    long row  = row0 + (lane & 15);
    long koff = k0 + (long)((lane >> 4) << 3);
    const __bf16* p = mat + row * ld + koff;
    bf16x8 lo = *(const bf16x8*)(p);
    bf16x8 hi = *(const bf16x8*)(p + 16);
    bf16x16 f;
#pragma unroll
    for (int i = 0; i < 8; ++i) { f[i] = lo[i]; f[i + 8] = hi[i]; }
    return f;
}

// B fragment (32x16 bf16, K x N) read from a row-major [N, K] matrix:
// lane holds column n=lane%16, K=[0..7]+[16..23] (lanes>=16: K=[8..15]+[24..31]).
static __device__ inline bf16x16 load_fragB(const __bf16* matNK, long ld,
                                            long n0, long k0, int lane) {
    long col  = n0 + (lane & 15);
    long koff = k0 + (long)((lane >> 4) << 3);
    const __bf16* p = matNK + col * ld + koff;
    bf16x8 lo = *(const bf16x8*)(p);
    bf16x8 hi = *(const bf16x8*)(p + 16);
    bf16x16 f;
#pragma unroll
    for (int i = 0; i < 8; ++i) { f[i] = lo[i]; f[i + 8] = hi[i]; }
    return f;
}

static __device__ inline f32x8 wmma_bf16(bf16x16 a, bf16x16 b, f32x8 c) {
    return __builtin_amdgcn_wmma_f32_16x16x32_bf16(false, a, false, b,
                                                   (short)0, c, false, false);
}

static __device__ inline float wave_sum(float v) {
#pragma unroll
    for (int off = 16; off > 0; off >>= 1) v += __shfl_xor(v, off, 32);
    return v;
}
static __device__ inline float wave_max(float v) {
#pragma unroll
    for (int off = 16; off > 0; off >>= 1) v = fmaxf(v, __shfl_xor(v, off, 32));
    return v;
}
static __device__ inline float wave_incl_scan(float v, int lane) {
#pragma unroll
    for (int off = 1; off < 32; off <<= 1) {
        float n = __shfl_up(v, off, 32);
        if (lane >= off) v += n;
    }
    return v;
}

// ---------------- K0: fp32 -> bf16 convert ----------------
__global__ void cvt_kernel(const float* __restrict__ src,
                           __bf16* __restrict__ dst, int n) {
    int i = blockIdx.x * blockDim.x + threadIdx.x;
    if (i < n) dst[i] = f2bf(src[i]);
}

// ---------------- K1: projection GEMM -------------------------------------
// C[4096,9216] = X16[4096,1024] @ Win16[9216,1024]^T + b_in
// B tile (64 cols x 32 K, 4 KB bf16) staged once per workgroup in LDS via
// CDNA5 async global->LDS loads, double-buffered, shared by all 8 waves.
// Output scattered to head-major bf16 slots: 9 slots of [BH][T][HD]
// (slot 2 = V stored transposed [BH][HD][T]). q/q_local pre-scaled by 0.125.
__global__ __launch_bounds__(256) void proj_gemm_kernel(
        const __bf16* __restrict__ x16, const __bf16* __restrict__ w16,
        const float* __restrict__ b_in, __bf16* __restrict__ heads) {
    __shared__ __bf16 Bsm[2][64 * 32];          // 2 x 4 KB double buffer

    const int tid  = threadIdx.x;
    const int lane = tid & 31;
    const int wave = tid >> 5;
    const long n0 = (long)blockIdx.x * 64;
    const long m0 = (long)blockIdx.y * 128 + (long)wave * 16;
    const long slot = (long)BH_ * T_ * HD_;

    // cooperative async stage: thread -> 16 bytes (8 bf16) of the B slice
    const int nl = tid >> 2;            // 0..63 local col
    const int k8 = (tid & 3) << 3;      // 0,8,16,24 within 32-K slice

    async_copy_b128(w16 + (n0 + nl) * E_ + k8, &Bsm[0][nl * 32 + k8]);
    wait_async0();
    __syncthreads();

    f32x8 acc0 = {}, acc1 = {}, acc2 = {}, acc3 = {};
#pragma unroll 2
    for (int kt = 0; kt < E_ / 32; ++kt) {
        const int cur = kt & 1;
        if (kt + 1 < E_ / 32)
            async_copy_b128(w16 + (n0 + nl) * E_ + (long)(kt + 1) * 32 + k8,
                            &Bsm[cur ^ 1][nl * 32 + k8]);
        long k0 = (long)kt * 32;
        if (kt + 1 < E_ / 32)
            __builtin_prefetch(x16 + (m0 + (lane & 15)) * E_ + k0 + 32, 0, 1);
        // hoist A + all 4 B fragments, then 4 back-to-back WMMAs
        bf16x16 a  = load_fragA(x16, E_, m0, k0, lane);
        bf16x16 b0 = load_fragB(&Bsm[cur][0], 32,  0, 0, lane);
        bf16x16 b1 = load_fragB(&Bsm[cur][0], 32, 16, 0, lane);
        bf16x16 b2 = load_fragB(&Bsm[cur][0], 32, 32, 0, lane);
        bf16x16 b3 = load_fragB(&Bsm[cur][0], 32, 48, 0, lane);
        acc0 = wmma_bf16(a, b0, acc0);
        acc1 = wmma_bf16(a, b1, acc1);
        acc2 = wmma_bf16(a, b2, acc2);
        acc3 = wmma_bf16(a, b3, acc3);
        wait_async0();
        __syncthreads();
    }

    const int roff = (lane >> 4) << 3;
    f32x8 accv[4] = {acc0, acc1, acc2, acc3};
#pragma unroll
    for (int nt = 0; nt < 4; ++nt) {
        long n = n0 + nt * 16 + (lane & 15);
        int w   = (int)(n / E_);
        int col = (int)(n % E_);
        int h = col >> 6, d = col & 63;
        float bias  = b_in[n];
        float scale = (w == 0 || w == 7) ? 0.125f : 1.0f;
        __bf16* sp = heads + (long)w * slot;
#pragma unroll
        for (int r = 0; r < 8; ++r) {
            long m = m0 + r + roff;
            int t = (int)(m >> 2), bb = (int)(m & 3);
            int bh = bb * H_ + h;
            float v = (accv[nt][r] + bias) * scale;
            long idx = (w == 2) ? (((long)bh * HD_ + d) * T_ + t)
                                : (((long)bh * T_ + t) * HD_ + d);
            sp[idx] = f2bf(v);
        }
    }
}

// ---------------- K2: fused attention --------------------------------------
// One workgroup (4 waves) per (bh, 16-row t-tile). LDS holds four 16x1024
// fp32 score tiles (left,right,global,local) = 256 KB (CDNA5: 320 KB/WGP).
// Mask "GEMMs" against triu become wave-level prefix scans.
__global__ __launch_bounds__(128) void attn_kernel(
        const __bf16* __restrict__ heads, __bf16* __restrict__ attn16) {
    extern __shared__ char smem_raw[];
    float*  S       = (float*)smem_raw;          // 4 x [16][1024] fp32
    __bf16* probs16 = (__bf16*)smem_raw;         // overlaps buffer 0 (reused)

    const int lane = threadIdx.x & 31;
    const int wave = threadIdx.x >> 5;
    const int bh   = blockIdx.x;
    const long t0  = (long)blockIdx.y * 16;
    const long slot = (long)BH_ * T_ * HD_;

    const int qs[4] = {3, 5, 0, 7};   // q_left, q_right, q, q_local
    const int ks[4] = {4, 6, 1, 8};   // k_left, k_right, k, k_local

    // ---- Phase 1: score tiles via WMMA (jobs = 4 types x 64 s-tiles) ----
    for (int job = wave; job < 256; job += 4) {
        int type  = job >> 6;
        long s0   = (long)(job & 63) * 16;
        const __bf16* qm = heads + (long)qs[type] * slot + (long)bh * T_ * HD_;
        const __bf16* km = heads + (long)ks[type] * slot + (long)bh * T_ * HD_;
        // hoist both k-step fragments, then 2 back-to-back WMMAs
        bf16x16 a0 = load_fragA(qm, HD_, t0, 0,  lane);
        bf16x16 a1 = load_fragA(qm, HD_, t0, 32, lane);
        bf16x16 b0 = load_fragB(km, HD_, s0, 0,  lane);
        bf16x16 b1 = load_fragB(km, HD_, s0, 32, lane);
        f32x8 c = {};
        c = wmma_bf16(a0, b0, c);
        c = wmma_bf16(a1, b1, c);
        float* buf = S + (long)type * 16 * 1024;
        int roff = (lane >> 4) << 3;
#pragma unroll
        for (int r = 0; r < 8; ++r)
            buf[(r + roff) * 1024 + s0 + (lane & 15)] = c[r];
    }
    __syncthreads();

    // ---- Phase 2+3: per-row softmax, then ordered prefix-scan masks ----
    float invden[4];
    for (int rr = 0; rr < 4; ++rr) {
        int row = wave * 4 + rr;
#pragma unroll
        for (int mat = 0; mat < 4; ++mat) {
            float* rp = S + (mat * 16 + row) * 1024;
            float mx = -3.0e38f;
            for (int i = lane; i < 1024; i += 32) mx = fmaxf(mx, rp[i]);
            mx = wave_max(mx);
            float sm = 0.f;
            for (int i = lane; i < 1024; i += 32) {
                float e = __expf(rp[i] - mx);
                rp[i] = e; sm += e;
            }
            sm = wave_sum(sm);
            float inv = 1.0f / sm;
            for (int i = lane; i < 1024; i += 32) rp[i] *= inv;
        }
        // ordered scan over s: cumulative masks replace sm@triu GEMMs
        float* L  = S + (0 * 16 + row) * 1024;
        float* R  = S + (1 * 16 + row) * 1024;
        float* Lo = S + (3 * 16 + row) * 1024;
        float cl = 0.f, cr = 0.f, den = 0.f;
        for (int ch = 0; ch < 32; ++ch) {
            int s = ch * 32 + lane;
            float ls = L[s], rs = R[s];
            float pl = cl + wave_incl_scan(ls, lane);  // prefix(left_sm)
            float pr = cr + wave_incl_scan(rs, lane);  // prefix(right_sm)
            float ncl = __shfl(pl, 31, 32);
            float ncr = __shfl(pr, 31, 32);
            // left_mask*right_mask + bw_left_mask*bw_right_mask
            float mask = pl * (1.f - pr + rs) + (1.f - pl + ls) * pr;
            float masked = Lo[s] * mask;
            Lo[s] = masked;
            den += masked;
            cl = ncl; cr = ncr;
        }
        den = wave_sum(den);
        invden[rr] = 1.0f / den;
    }
    __syncthreads();

    // ---- Phase 3b: combined probs -> bf16 in LDS (reuses left buffer) ----
    for (int rr = 0; rr < 4; ++rr) {
        int row = wave * 4 + rr;
        float* G  = S + (2 * 16 + row) * 1024;
        float* Lo = S + (3 * 16 + row) * 1024;
        float inv = invden[rr];
        for (int i = lane; i < 1024; i += 32) {
            float comb = 0.5f * (G[i] + Lo[i] * inv);
            probs16[row * 1024 + i] = f2bf(comb);
        }
    }
    __syncthreads();

    // ---- Phase 4: attn tile = probs @ V (V pre-transposed [bh][hd][s]) ----
    const __bf16* vt = heads + 2 * slot + (long)bh * HD_ * T_;
    const int n0 = wave * 16;
    f32x8 acc = {};
#pragma unroll 2
    for (int kt = 0; kt < 32; ++kt) {
        long k0 = (long)kt * 32;
        bf16x16 a = load_fragA(probs16, 1024, 0, k0, lane);   // LDS
        bf16x16 b = load_fragB(vt, T_, n0, k0, lane);         // global
        acc = wmma_bf16(a, b, acc);
    }
    const int roff = (lane >> 4) << 3;
    const int b_ = bh >> 4, h = bh & 15;
#pragma unroll
    for (int r = 0; r < 8; ++r) {
        long t = t0 + r + roff;
        int  e = h * 64 + n0 + (lane & 15);
        attn16[(t * B_ + b_) * (long)E_ + e] = f2bf(acc[r]);
    }
}

// ---------------- K3: output GEMM ------------------------------------------
// out[4096,1024] = attn16 @ Wout16[1024,1024]^T + b_out  (fp32 out)
// Same async-LDS B staging as K1.
__global__ __launch_bounds__(256) void out_gemm_kernel(
        const __bf16* __restrict__ a16, const __bf16* __restrict__ w16,
        const float* __restrict__ bias, float* __restrict__ out) {
    __shared__ __bf16 Bsm[2][64 * 32];

    const int tid  = threadIdx.x;
    const int lane = tid & 31;
    const int wave = tid >> 5;
    const long n0 = (long)blockIdx.x * 64;
    const long m0 = (long)blockIdx.y * 128 + (long)wave * 16;

    const int nl = tid >> 2;
    const int k8 = (tid & 3) << 3;

    async_copy_b128(w16 + (n0 + nl) * E_ + k8, &Bsm[0][nl * 32 + k8]);
    wait_async0();
    __syncthreads();

    f32x8 acc0 = {}, acc1 = {}, acc2 = {}, acc3 = {};
#pragma unroll 2
    for (int kt = 0; kt < E_ / 32; ++kt) {
        const int cur = kt & 1;
        if (kt + 1 < E_ / 32)
            async_copy_b128(w16 + (n0 + nl) * E_ + (long)(kt + 1) * 32 + k8,
                            &Bsm[cur ^ 1][nl * 32 + k8]);
        long k0 = (long)kt * 32;
        if (kt + 1 < E_ / 32)
            __builtin_prefetch(a16 + (m0 + (lane & 15)) * E_ + k0 + 32, 0, 1);
        bf16x16 a  = load_fragA(a16, E_, m0, k0, lane);
        bf16x16 b0 = load_fragB(&Bsm[cur][0], 32,  0, 0, lane);
        bf16x16 b1 = load_fragB(&Bsm[cur][0], 32, 16, 0, lane);
        bf16x16 b2 = load_fragB(&Bsm[cur][0], 32, 32, 0, lane);
        bf16x16 b3 = load_fragB(&Bsm[cur][0], 32, 48, 0, lane);
        acc0 = wmma_bf16(a, b0, acc0);
        acc1 = wmma_bf16(a, b1, acc1);
        acc2 = wmma_bf16(a, b2, acc2);
        acc3 = wmma_bf16(a, b3, acc3);
        wait_async0();
        __syncthreads();
    }

    const int roff = (lane >> 4) << 3;
    f32x8 accv[4] = {acc0, acc1, acc2, acc3};
#pragma unroll
    for (int nt = 0; nt < 4; ++nt) {
        long n = n0 + nt * 16 + (lane & 15);
        float bv = bias[n];
#pragma unroll
        for (int r = 0; r < 8; ++r) {
            long m = m0 + r + roff;
            out[m * E_ + n] = accv[nt][r] + bv;
        }
    }
}

// ---------------- host launch ----------------------------------------------
extern "C" void kernel_launch(void* const* d_in, const int* in_sizes, int n_in,
                              void* d_out, int out_size, void* d_ws, size_t ws_size,
                              hipStream_t stream) {
    const float* x     = (const float*)d_in[0];
    const float* w_in  = (const float*)d_in[1];
    const float* b_in  = (const float*)d_in[2];
    const float* w_out = (const float*)d_in[3];
    const float* b_out = (const float*)d_in[4];
    float* out = (float*)d_out;

    char* ws = (char*)d_ws;
    __bf16* x16    = (__bf16*)(ws);                         //  8.0 MB
    __bf16* win16  = (__bf16*)(ws + 8388608);               // 18.0 MB
    __bf16* wout16 = (__bf16*)(ws + 27262976);              //  2.0 MB
    __bf16* heads  = (__bf16*)(ws + 29360128);              // 9 x 8 MB
    __bf16* attn16 = (__bf16*)(ws + 104857600);             //  8.0 MB

    // K0: converts
    cvt_kernel<<<(MROWS * E_ + 255) / 256, 256, 0, stream>>>(x, x16, MROWS * E_);
    cvt_kernel<<<(NPROJ * E_ + 255) / 256, 256, 0, stream>>>(w_in, win16, NPROJ * E_);
    cvt_kernel<<<(E_ * E_ + 255) / 256, 256, 0, stream>>>(w_out, wout16, E_ * E_);

    // K1: projection GEMM (N=9216 -> 144 col-blocks of 64, M=4096 -> 32 row-blocks of 128)
    proj_gemm_kernel<<<dim3(144, 32), 256, 0, stream>>>(x16, win16, b_in, heads);

    // K2: fused attention, 256 KB dynamic LDS per workgroup
    attn_kernel<<<dim3(BH_, T_ / 16), 128, 4 * 16 * 1024 * sizeof(float), stream>>>(heads, attn16);

    // K3: output GEMM
    out_gemm_kernel<<<dim3(16, 32), 256, 0, stream>>>(attn16, wout16, b_out, out);
}
